// DProdQ_2448131359012
// MI455X (gfx1250) — matrix operand order repaired and checked by reading, not compiled, not run on hardware
//
#include <hip/hip_runtime.h>

// Problem constants (from reference): N=131072, D=512, M=4, K=256, DM=128
#define N_TOTAL 131072
#define D_DIM   512
#define M_SUB   4
#define K_CODES 256
#define DM_SUB  128

typedef __attribute__((ext_vector_type(16))) __bf16 v16bf;
typedef __attribute__((ext_vector_type(8)))  float  v8f;
typedef __attribute__((ext_vector_type(4)))  unsigned u32x4;
typedef __attribute__((ext_vector_type(8)))  unsigned u32x8;

__device__ __forceinline__ unsigned short f32_to_bf16_bits(float f) {
  unsigned u = __builtin_bit_cast(unsigned, f);
  u += 0x7FFFu + ((u >> 16) & 1u);   // round-to-nearest-even
  return (unsigned short)(u >> 16);
}
__device__ __forceinline__ __bf16 bits_to_bf16(unsigned short s) {
  return __builtin_bit_cast(__bf16, s);
}

__device__ __forceinline__ v8f wmma_bf16(v16bf a, v16bf b, v8f c) {
  // D = A(16x32 bf16) * B(32x16 bf16) + C(16x16 f32)
  return __builtin_amdgcn_wmma_f32_16x16x32_bf16(false, a, false, b, (short)0, c,
                                                 false, false);
}

// ---- Fragment loaders (wave32): contiguous k order k = 16*(lane/16) + s, same for A and B ----

// A-matrix 16x32 bf16: lane<16 holds row=lane, K 0..15; lane>=16 holds
// row=lane-16, K 16..31; 2 bf16 per VGPR.
__device__ __forceinline__ v16bf load_a_from_f32(const float* src, int stride, int lane) {
  const int r  = lane & 15;
  const int kb = (lane < 16) ? 0 : 16;
  const float* p = src + r * stride + kb;
  v16bf a;
#pragma unroll
  for (int e = 0; e < 8; ++e) a[e] = bits_to_bf16(f32_to_bf16_bits(p[e]));
#pragma unroll
  for (int e = 0; e < 8; ++e) a[8 + e] = bits_to_bf16(f32_to_bf16_bits(p[8 + e]));
  return a;
}

__device__ __forceinline__ v16bf load_a_from_bf(const unsigned short* src, int stride, int lane) {
  const int r  = lane & 15;
  const int kb = (lane < 16) ? 0 : 16;
  const unsigned short* p = src + r * stride + kb;
  v16bf a;
#pragma unroll
  for (int e = 0; e < 8; ++e) a[e] = bits_to_bf16(p[e]);
#pragma unroll
  for (int e = 0; e < 8; ++e) a[8 + e] = bits_to_bf16(p[8 + e]);
  return a;
}

// B-matrix 32x16 bf16: lane<16 -> col=lane, K 0..15; lane>=16 -> col=lane-16,
// K 16..31; VGPR v packs K=2v,2v+1 of the lane's half. Memory is "T" layout:
// srcT[col*strideT + k] == B[k][col]  ->  packed pairs are one dword load.
__device__ __forceinline__ v16bf load_b_from_t(const unsigned short* srcT, int strideT, int lane) {
  const int c  = lane & 15;
  const int kb = (lane < 16) ? 0 : 16;
  const unsigned short* p = srcT + c * strideT + kb;
  v16bf b;
#pragma unroll
  for (int v = 0; v < 8; ++v) {
    const unsigned w = *(const unsigned*)(p + 2 * v);
    b[2 * v]     = bits_to_bf16((unsigned short)(w & 0xFFFFu));
    b[2 * v + 1] = bits_to_bf16((unsigned short)(w >> 16));
  }
  return b;
}

// ---------------- prep: bf16 conversions, c2, accumulator zeroing -----------
__global__ void dprodq_prep_kernel(const float* __restrict__ codebook,
                                   const float* __restrict__ rotate,
                                   unsigned short* __restrict__ rotT,
                                   unsigned short* __restrict__ cb_bf,
                                   unsigned short* __restrict__ cbT_bf,
                                   float* __restrict__ c2,
                                   float* __restrict__ accum) {
  const int tid = blockIdx.x * blockDim.x + threadIdx.x;
  const int nth = gridDim.x * blockDim.x;

  // rotT[n][k] = rotate[k][n]  (B of stage-1 GEMM, K-contiguous)
  for (int i = tid; i < D_DIM * D_DIM; i += nth) {
    const int n = i >> 9, k = i & (D_DIM - 1);
    rotT[i] = f32_to_bf16_bits(rotate[(size_t)k * D_DIM + n]);
  }
  // cb_bf row-major [m][k][d]; cbT_bf transposed [m][d][k]
  for (int i = tid; i < M_SUB * K_CODES * DM_SUB; i += nth) {
    const float v = codebook[i];
    const unsigned short bv = f32_to_bf16_bits(v);
    cb_bf[i] = bv;
    const int d = i & (DM_SUB - 1);
    const int k = (i >> 7) & (K_CODES - 1);
    const int m = i >> 15;
    cbT_bf[((size_t)m * DM_SUB + d) * K_CODES + k] = bv;
  }
  // c2[m][k] = sum_d codebook^2
  for (int i = tid; i < M_SUB * K_CODES; i += nth) {
    const float* row = codebook + (size_t)i * DM_SUB;
    float s = 0.0f;
    for (int d = 0; d < DM_SUB; ++d) s = fmaf(row[d], row[d], s);
    c2[i] = s;
  }
  if (tid < 4) accum[tid] = 0.0f;   // soft, hard, joint, reg sums
}

// ---------------- fused main kernel: 32 rows of x per block -----------------
// LDS map (dynamic):
//   [  0KB..64KB )  x tile f32 (TDM destination)
//   [ 64KB..96KB )  x tile bf16
//   [ 96KB..160KB)  xr f32
//   [160KB..288KB)  xc -> probs f32
//   [288KB..+   )   x2 / codes / reduction scratch
__launch_bounds__(256, 1)
__global__ void dprodq_main_kernel(const float* __restrict__ x,
                                   const float* __restrict__ codebook,
                                   const unsigned short* __restrict__ rotT,
                                   const unsigned short* __restrict__ cb_bf,
                                   const unsigned short* __restrict__ cbT_bf,
                                   const float* __restrict__ c2,
                                   float* __restrict__ codes_out,
                                   float* __restrict__ accum) {
  extern __shared__ unsigned char smem[];
  float*          lds_xf  = (float*)smem;                           // 64 KB (TDM dst)
  unsigned short* lds_xbf = (unsigned short*)(smem + 64 * 1024);    // 32 KB
  float*          lds_xr  = (float*)(smem + 96 * 1024);             // 64 KB
  float*          lds_xc  = (float*)(smem + 160 * 1024);            // 128 KB
  float*          lds_x2  = (float*)(smem + 288 * 1024);            // 512 B
  int*            lds_code= (int*)  (smem + 288 * 1024 + 512);      // 512 B
  float*          lds_red = (float*)(smem + 288 * 1024 + 1024);     // 12 B

  const int tid  = threadIdx.x;
  const int lane = tid & 31;
  const int wave = tid >> 5;
  const int row0 = blockIdx.x * 32;

  // Stage -1: Tensor Data Mover DMA of the x tile (32 x 512 f32) into LDS.
  // D# per CDNA5 ISA (2D tensor, 2 SGPR groups).
  if (wave == 0) {
    const unsigned long long ga =
        (unsigned long long)(const void*)(x + (size_t)row0 * D_DIM);
    u32x4 g0;
    g0[0] = 1u;                                        // count=1, user mode
    g0[1] = 0u;                                        // lds_addr = 0 bytes
    g0[2] = (unsigned)(ga & 0xFFFFFFFFu);              // global_addr[31:0]
    g0[3] = (unsigned)((ga >> 32) & 0x01FFFFFFu)       // global_addr[56:32]
            | (2u << 30);                              // type = 2 ("image")
    u32x8 g1;
    g1[0] = (2u << 16);                                // data_size = 4 bytes
    g1[1] = ((unsigned)D_DIM & 0xFFFFu) << 16;         // tensor_dim0[15:0]
    g1[2] = (((unsigned)D_DIM >> 16) & 0xFFFFu)        // tensor_dim0[31:16]
            | (((unsigned)N_TOTAL & 0xFFFFu) << 16);   // tensor_dim1[15:0]
    g1[3] = (((unsigned)N_TOTAL >> 16) & 0xFFFFu)      // tensor_dim1[31:16]
            | ((unsigned)D_DIM << 16);                 // tile_dim0 = 512
    g1[4] = 32u;                                       // tile_dim1 = 32, tile_dim2 = 0
    g1[5] = (unsigned)D_DIM;                           // tensor_dim0_stride[31:0]
    g1[6] = 0u;                                        // stride0[47:32], stride1[15:0]
    g1[7] = 0u;
    asm volatile("tensor_load_to_lds %0, %1" :: "s"(g0), "s"(g1) : "memory");
    __builtin_amdgcn_s_wait_tensorcnt(0);              // s_wait_tensorcnt 0x0
  }
  if (tid < 3) lds_red[tid] = 0.0f;
  __builtin_prefetch(rotT + (size_t)(wave * 64) * D_DIM, 0, 1);  // global_prefetch_b8
  __syncthreads();

  // Stage 0: convert LDS f32 tile -> LDS bf16 tile
  for (int i = tid; i < 32 * D_DIM; i += 256) {
    lds_xbf[i] = f32_to_bf16_bits(lds_xf[i]);
  }
  __syncthreads();

  // Stage 1: xr = x_bf16 @ rotate_bf16  (wave w -> columns [64w, 64w+64))
  {
    const int col0 = wave * 64;
    v8f acc[2][4];
#pragma unroll
    for (int rt = 0; rt < 2; ++rt)
#pragma unroll
      for (int ct = 0; ct < 4; ++ct)
#pragma unroll
        for (int e = 0; e < 8; ++e) acc[rt][ct][e] = 0.0f;

    for (int kb = 0; kb < 16; ++kb) {           // K = 512, step 32
      const int k0 = kb * 32;
      const v16bf a0 = load_a_from_bf(lds_xbf + k0, D_DIM, lane);
      const v16bf a1 = load_a_from_bf(lds_xbf + 16 * D_DIM + k0, D_DIM, lane);
#pragma unroll
      for (int ct = 0; ct < 4; ++ct) {
        const v16bf b = load_b_from_t(rotT + (size_t)(col0 + ct * 16) * D_DIM + k0, D_DIM, lane);
        acc[0][ct] = wmma_bf16(a0, b, acc[0][ct]);
        acc[1][ct] = wmma_bf16(a1, b, acc[1][ct]);
      }
    }
    const int rlo = (lane < 16) ? 0 : 8;
    const int cc  = lane & 15;
#pragma unroll
    for (int rt = 0; rt < 2; ++rt)
#pragma unroll
      for (int ct = 0; ct < 4; ++ct)
#pragma unroll
        for (int v = 0; v < 8; ++v)
          lds_xr[(rt * 16 + rlo + v) * D_DIM + col0 + ct * 16 + cc] = acc[rt][ct][v];
  }
  __syncthreads();

  // Stage 2: x2[n][m] = ||splits||^2
  if (tid < 128) {
    const int n = tid & 31, m = tid >> 5;
    const float* pr = lds_xr + n * D_DIM + m * DM_SUB;
    float s = 0.0f;
    for (int d = 0; d < DM_SUB; ++d) s = fmaf(pr[d], pr[d], s);
    lds_x2[m * 32 + n] = s;
  }
  __syncthreads();

  // Stage 3: xc = splits @ codebook^T  (wave w -> (m=w>>1, k-half=w&1))
  {
    const int m = wave >> 1;
    const int kbase = (wave & 1) * 128;
    v8f acc[2][8];
#pragma unroll
    for (int rt = 0; rt < 2; ++rt)
#pragma unroll
      for (int ct = 0; ct < 8; ++ct)
#pragma unroll
        for (int e = 0; e < 8; ++e) acc[rt][ct][e] = 0.0f;

#pragma unroll
    for (int dk = 0; dk < 4; ++dk) {            // K = DM = 128, step 32
      const int d0 = dk * 32;
      const v16bf a0 = load_a_from_f32(lds_xr + m * DM_SUB + d0, D_DIM, lane);
      const v16bf a1 = load_a_from_f32(lds_xr + 16 * D_DIM + m * DM_SUB + d0, D_DIM, lane);
#pragma unroll
      for (int ct = 0; ct < 8; ++ct) {
        const v16bf b = load_b_from_t(
            cb_bf + (size_t)(m * K_CODES + kbase + ct * 16) * DM_SUB + d0, DM_SUB, lane);
        acc[0][ct] = wmma_bf16(a0, b, acc[0][ct]);
        acc[1][ct] = wmma_bf16(a1, b, acc[1][ct]);
      }
    }
    const int rlo = (lane < 16) ? 0 : 8;
    const int cc  = lane & 15;
#pragma unroll
    for (int rt = 0; rt < 2; ++rt)
#pragma unroll
      for (int ct = 0; ct < 8; ++ct)
#pragma unroll
        for (int v = 0; v < 8; ++v) {
          const int n = rt * 16 + rlo + v;
          lds_xc[(m * 32 + n) * K_CODES + kbase + ct * 16 + cc] = acc[rt][ct][v];
        }
  }
  __syncthreads();

  // Stage 4: logits = 2*xc - x2 - c2 ; softmax (in place) + argmax
  if (tid < 128) {
    const int n = tid & 31, m = tid >> 5;
    float* pxc = lds_xc + (m * 32 + n) * K_CODES;
    const float* pc2 = c2 + m * K_CODES;
    const float x2v = lds_x2[m * 32 + n];
    float best = -3.4e38f; int bi = 0;
    for (int k = 0; k < K_CODES; ++k) {
      const float lg = 2.0f * pxc[k] - x2v - pc2[k];
      pxc[k] = lg;
      if (lg > best) { best = lg; bi = k; }
    }
    float ssum = 0.0f;
    for (int k = 0; k < K_CODES; ++k) { const float e = __expf(pxc[k] - best); pxc[k] = e; ssum += e; }
    const float inv = 1.0f / ssum;
    for (int k = 0; k < K_CODES; ++k) pxc[k] *= inv;
    lds_code[n * 4 + m] = bi;
    codes_out[(size_t)(row0 + n) * M_SUB + m] = (float)bi;
  }
  __syncthreads();

  // Stage 5: soft = probs @ codebook ; accumulate soft/hard/joint SSE
  {
    const int m = wave >> 1;
    const int dbase = (wave & 1) * 64;
    v8f acc[2][4];
#pragma unroll
    for (int rt = 0; rt < 2; ++rt)
#pragma unroll
      for (int ct = 0; ct < 4; ++ct)
#pragma unroll
        for (int e = 0; e < 8; ++e) acc[rt][ct][e] = 0.0f;

    for (int kb = 0; kb < 8; ++kb) {            // K = 256, step 32
      const int k0 = kb * 32;
      const v16bf a0 = load_a_from_f32(lds_xc + (m * 32) * K_CODES + k0, K_CODES, lane);
      const v16bf a1 = load_a_from_f32(lds_xc + (m * 32 + 16) * K_CODES + k0, K_CODES, lane);
#pragma unroll
      for (int ct = 0; ct < 4; ++ct) {
        const v16bf b = load_b_from_t(
            cbT_bf + (size_t)(m * DM_SUB + dbase + ct * 16) * K_CODES + k0, K_CODES, lane);
        acc[0][ct] = wmma_bf16(a0, b, acc[0][ct]);
        acc[1][ct] = wmma_bf16(a1, b, acc[1][ct]);
      }
    }

    float s_soft = 0.0f, s_hard = 0.0f, s_joint = 0.0f;
    const int rlo = (lane < 16) ? 0 : 8;
    const int cc  = lane & 15;
#pragma unroll
    for (int rt = 0; rt < 2; ++rt)
#pragma unroll
      for (int ct = 0; ct < 4; ++ct)
#pragma unroll
        for (int v = 0; v < 8; ++v) {
          const int n = rt * 16 + rlo + v;
          const int d = dbase + ct * 16 + cc;
          const float softv  = acc[rt][ct][v];
          const float splitv = lds_xr[n * D_DIM + m * DM_SUB + d];
          const int   code   = lds_code[n * 4 + m];
          const float hardv  = codebook[(size_t)(m * K_CODES + code) * DM_SUB + d];
          float e;
          e = splitv - softv; s_soft  = fmaf(e, e, s_soft);
          e = splitv - hardv; s_hard  = fmaf(e, e, s_hard);
          e = softv  - hardv; s_joint = fmaf(e, e, s_joint);
        }
#pragma unroll
    for (int off = 16; off > 0; off >>= 1) {    // wave32 reduction
      s_soft  += __shfl_xor(s_soft,  off, 32);
      s_hard  += __shfl_xor(s_hard,  off, 32);
      s_joint += __shfl_xor(s_joint, off, 32);
    }
    if (lane == 0) {
      atomicAdd(&lds_red[0], s_soft);
      atomicAdd(&lds_red[1], s_hard);
      atomicAdd(&lds_red[2], s_joint);
    }
  }
  __syncthreads();
  if (tid < 3) atomicAdd(&accum[tid], lds_red[tid]);
}

// ---------------- reg = mean((R R^T - I)^2) --------------------------------
__global__ void dprodq_reg_kernel(const float* __restrict__ rotate,
                                  float* __restrict__ accum) {
  const int idx = blockIdx.x * blockDim.x + threadIdx.x;   // 0 .. 512*512-1
  const int i = idx >> 9, j = idx & (D_DIM - 1);
  const float* ri = rotate + (size_t)i * D_DIM;
  const float* rj = rotate + (size_t)j * D_DIM;
  float s = 0.0f;
  for (int k = 0; k < D_DIM; ++k) s = fmaf(ri[k], rj[k], s);
  const float dlt = s - ((i == j) ? 1.0f : 0.0f);
  float v = dlt * dlt;
#pragma unroll
  for (int off = 16; off > 0; off >>= 1) v += __shfl_xor(v, off, 32);
  __shared__ float red;
  if (threadIdx.x == 0) red = 0.0f;
  __syncthreads();
  if ((threadIdx.x & 31) == 0) atomicAdd(&red, v);
  __syncthreads();
  if (threadIdx.x == 0) atomicAdd(&accum[3], red);
}

// ---------------- finalize loss --------------------------------------------
__global__ void dprodq_final_kernel(const float* __restrict__ accum,
                                    float* __restrict__ out_loss) {
  const float inv_nd = 1.0f / ((float)N_TOTAL * (float)DM_SUB);
  const float soft_d = accum[0] * inv_nd;
  const float hard_d = accum[1] * inv_nd;
  const float joint  = accum[2] * inv_nd;
  const float reg    = accum[3] / ((float)D_DIM * (float)D_DIM);
  out_loss[0] = 0.1f * soft_d + hard_d + 0.1f * joint + 0.01f * reg;
}

extern "C" void kernel_launch(void* const* d_in, const int* in_sizes, int n_in,
                              void* d_out, int out_size, void* d_ws, size_t ws_size,
                              hipStream_t stream) {
  const float* x        = (const float*)d_in[0];   // (N, D)
  const float* codebook = (const float*)d_in[1];   // (M, K, DM)
  const float* rotate   = (const float*)d_in[2];   // (D, D)
  float* out = (float*)d_out;                      // codes (N*M) then loss

  unsigned char* ws = (unsigned char*)d_ws;
  unsigned short* rotT   = (unsigned short*)ws;                    // 512 KB
  unsigned short* cb_bf  = (unsigned short*)(ws + 512 * 1024);     // 256 KB
  unsigned short* cbT_bf = (unsigned short*)(ws + 768 * 1024);     // 256 KB
  float*          c2     = (float*)(ws + 1024 * 1024);             // 4 KB
  float*          accum  = (float*)(ws + 1028 * 1024);             // 16 B

  dprodq_prep_kernel<<<256, 256, 0, stream>>>(codebook, rotate, rotT, cb_bf, cbT_bf, c2, accum);

  const size_t smem_bytes = 290 * 1024;   // 288KB tiles + small scratch
  dprodq_main_kernel<<<N_TOTAL / 32, 256, smem_bytes, stream>>>(
      x, codebook, rotT, cb_bf, cbT_bf, c2, out, accum);

  dprodq_reg_kernel<<<(D_DIM * D_DIM) / 256, 256, 0, stream>>>(rotate, accum);

  dprodq_final_kernel<<<1, 1, 0, stream>>>(accum, out + (size_t)N_TOTAL * M_SUB);
}
